// VectorQuantizer_23424751632461
// MI455X (gfx1250) — matrix operand order repaired and checked by reading, not compiled, not run on hardware
//
#include <hip/hip_runtime.h>
#include <hip/hip_bf16.h>
#include <math.h>

typedef __attribute__((ext_vector_type(16))) __bf16 v16bf;
typedef __attribute__((ext_vector_type(8)))  float  v8f;
typedef __attribute__((ext_vector_type(4)))  float  v4f;

#define N_PTS   65536      // 16*64*64 points
#define N_EMB   1024
#define EMB_DIM 64
#define IMG_PIX 4096       // 64*64

// workspace byte offsets
#define WS_WB    0u         // 1024*64 bf16  = 131072 B
#define WS_BIAS  131072u    // 1024 f32      =   4096 B
#define WS_IDX   135168u    // 65536 int     = 262144 B
#define WS_HIST  397312u    // 1024 uint     =   4096 B
#define WS_ERRP  401408u    // 4096 f32      =  16384 B

// ---------------------------------------------------------------- prep ----
__global__ void __launch_bounds__(256)
vq_prep(const float* __restrict__ W, __bf16* __restrict__ Wb,
        float* __restrict__ bias, unsigned* __restrict__ hist)
{
    int k = blockIdx.x * blockDim.x + threadIdx.x;   // 0..1023 (grid 4x256)
    if (k < N_EMB) {
        float s = 0.f;
        for (int d = 0; d < EMB_DIM; ++d) {
            float w = W[k * EMB_DIM + d];
            Wb[k * EMB_DIM + d] = (__bf16)w;
            s += w * w;
        }
        bias[k] = 0.5f * s;
        hist[k] = 0u;
    }
}

// -------------------------------------------------------------- assign ----
// One wave32 handles 16 consecutive points. bf16 WMMA over the 1024-code
// dimension in 64 chunks of 16 codes; K = 64 split into two 32-wide WMMAs.
__global__ void __launch_bounds__(256)
vq_assign(const float* __restrict__ x, const float* __restrict__ W,
          const __bf16* __restrict__ Wb, const float* __restrict__ bias,
          int* __restrict__ idxOut, unsigned* __restrict__ hist,
          float* __restrict__ errPart, float* __restrict__ quantOut)
{
    __shared__ int sIdx[8 * 16];

    const int wave = threadIdx.x >> 5;
    const int lane = threadIdx.x & 31;
    const int hf   = lane >> 4;        // half of wave (0/1)
    const int l16  = lane & 15;
    const int rowBase = (blockIdx.x * 8 + wave) * 16;   // first point id
    const int b       = rowBase >> 12;                  // batch image
    const int pixBase = rowBase & (IMG_PIX - 1);        // pixel within image
    const float* __restrict__ xb = x + (size_t)b * (EMB_DIM * IMG_PIX);

    // ---- A fragments: 16 points x 64 dims as two 16x32 bf16 tiles.
    // 16-bit A layout (matches B): lane<16 holds K 0..15, lane>=16 holds K 16..31.
    v16bf a0, a1;
    #pragma unroll
    for (int e = 0; e < 16; ++e) {
        const int kk = e + (hf << 4);   // K within 32-chunk
        a0[e] = (__bf16)xb[(size_t)(kk)      * IMG_PIX + pixBase + l16];
        a1[e] = (__bf16)xb[(size_t)(kk + 32) * IMG_PIX + pixBase + l16];
    }

    // ---- running argmax of (x . w_k - 0.5||w_k||^2) over the code axis
    float bestS[8];
    int   bestI[8];
    #pragma unroll
    for (int r = 0; r < 8; ++r) { bestS[r] = -3.0e38f; bestI[r] = 0; }

    for (int c = 0; c < N_EMB / 16; ++c) {
        const int code = c * 16 + l16;                 // N = lane&15
        // B layout used here (same as A): lane element e <-> K = e + 16*hf, N = lane&15
        const __bf16* wrow = Wb + (size_t)code * EMB_DIM + hf * 16;
        v16bf b0 = *(const v16bf*)(wrow);              // K-chunk 0 (d 0..31)
        v16bf b1 = *(const v16bf*)(wrow + 32);         // K-chunk 1 (d 32..63)
        const float bb = bias[code];

        v8f acc = {};
        acc = __builtin_amdgcn_wmma_f32_16x16x32_bf16(false, a0, false, b0,
                                                      (short)0, acc, false, false);
        acc = __builtin_amdgcn_wmma_f32_16x16x32_bf16(false, a1, false, b1,
                                                      (short)0, acc, false, false);
        // C/D layout: reg r, this lane <-> point M = r + 8*hf, code N = l16
        #pragma unroll
        for (int r = 0; r < 8; ++r) {
            float s = acc[r] - bb;
            if (s > bestS[r]) { bestS[r] = s; bestI[r] = code; }
        }
    }

    // ---- butterfly reduce over the 16 lanes of each half (distinct code%16);
    // tie-break on smaller index to match argmin-first semantics.
    #pragma unroll
    for (int r = 0; r < 8; ++r) {
        float s = bestS[r]; int i = bestI[r];
        #pragma unroll
        for (int mask = 8; mask >= 1; mask >>= 1) {
            float s2 = __shfl_xor(s, mask, 32);
            int   i2 = __shfl_xor(i, mask, 32);
            if (s2 > s || (s2 == s && i2 < i)) { s = s2; i = i2; }
        }
        if (l16 == 0) sIdx[wave * 16 + r + 8 * hf] = i;   // point M = r+8*hf
    }
    // same-wave LDS round trip; compiler inserts ds waits
    const int myIdx = sIdx[wave * 16 + l16];              // idx of point l16
    if (lane < 16) {
        idxOut[rowBase + l16] = myIdx;
        atomicAdd(&hist[myIdx], 1u);
    }

    // ---- gather W[idx] -> quantized (NCHW) + squared-error partials.
    // lanes 0-15 do even dims of points 0-15, lanes 16-31 odd dims.
    const float* __restrict__ wq = W + (size_t)myIdx * EMB_DIM;
    float* __restrict__ qb = quantOut + (size_t)b * (EMB_DIM * IMG_PIX);
    float sum = 0.f;
    #pragma unroll 8
    for (int it = 0; it < 32; ++it) {
        const int d = it * 2 + hf;
        const size_t off = (size_t)d * IMG_PIX + pixBase + l16;
        const float q  = wq[d];
        const float xv = xb[off];
        __builtin_nontemporal_store(q, qb + off);
        const float e = q - xv;
        sum += e * e;
    }
    #pragma unroll
    for (int mask = 16; mask >= 1; mask >>= 1)
        sum += __shfl_xor(sum, mask, 32);
    if (lane == 0) errPart[blockIdx.x * 8 + wave] = sum;  // deterministic
}

// ----------------------------------------------------------- encodings ----
// One wave per row: 1024 floats as 8 contiguous 128-float bursts (v4f NT).
__global__ void __launch_bounds__(256)
vq_encodings(const int* __restrict__ idxArr, float* __restrict__ enc)
{
    const int wave = threadIdx.x >> 5;
    const int lane = threadIdx.x & 31;
    const int row  = blockIdx.x * 8 + wave;
    const int idx  = idxArr[row];
    float* __restrict__ rp = enc + (size_t)row * N_EMB;
    #pragma unroll
    for (int j = 0; j < 8; ++j) {
        const int col = j * 128 + lane * 4;
        v4f v = {0.f, 0.f, 0.f, 0.f};
        const int rel = idx - col;
        if (rel >= 0 && rel < 4) v[rel] = 1.0f;
        __builtin_nontemporal_store(v, (v4f*)(rp + col));
    }
}

// ------------------------------------------------------------ finalize ----
__global__ void __launch_bounds__(1024)
vq_final(const float* __restrict__ errPart, const unsigned* __restrict__ hist,
         float* __restrict__ outLoss, float* __restrict__ outPerp)
{
    __shared__ float red[1024];
    const int t = threadIdx.x;

    // deterministic loss reduction over 4096 wave partials
    float s = errPart[t] + errPart[t + 1024] + errPart[t + 2048] + errPart[t + 3072];
    red[t] = s; __syncthreads();
    for (int o = 512; o > 0; o >>= 1) {
        if (t < o) red[t] += red[t + o];
        __syncthreads();
    }
    const float lossSum = red[0];
    __syncthreads();

    // perplexity
    const float p = (float)hist[t] * (1.0f / 65536.0f);
    red[t] = p * logf(p + 1e-10f);
    __syncthreads();
    for (int o = 512; o > 0; o >>= 1) {
        if (t < o) red[t] += red[t + o];
        __syncthreads();
    }
    if (t == 0) {
        *outLoss = 1.25f * lossSum / 4194304.0f;   // (1+0.25)*MSE, N*D = 4194304
        *outPerp = expf(-red[0]);
    }
}

// -------------------------------------------------------------- launch ----
extern "C" void kernel_launch(void* const* d_in, const int* in_sizes, int n_in,
                              void* d_out, int out_size, void* d_ws, size_t ws_size,
                              hipStream_t stream)
{
    (void)in_sizes; (void)n_in; (void)out_size; (void)ws_size;
    const float* x = (const float*)d_in[0];   // [16,64,64,64] NCHW
    const float* W = (const float*)d_in[1];   // [1024,64]

    char* ws = (char*)d_ws;
    __bf16*   Wb      = (__bf16*)(ws + WS_WB);
    float*    bias    = (float*)(ws + WS_BIAS);
    int*      idxArr  = (int*)(ws + WS_IDX);
    unsigned* hist    = (unsigned*)(ws + WS_HIST);
    float*    errPart = (float*)(ws + WS_ERRP);

    float* out      = (float*)d_out;
    float* outLoss  = out;                       // [0]
    float* quantOut = out + 1;                   // 4,194,304 floats (NCHW)
    float* outPerp  = out + 1 + 4194304;         // [4194305]
    float* enc      = out + 4194306;             // 65536*1024 floats

    vq_prep     <<<4,    256, 0, stream>>>(W, Wb, bias, hist);
    vq_assign   <<<512,  256, 0, stream>>>(x, W, Wb, bias, idxArr, hist, errPart, quantOut);
    vq_encodings<<<8192, 256, 0, stream>>>(idxArr, enc);
    vq_final    <<<1,   1024, 0, stream>>>(errPart, hist, outLoss, outPerp);
}
